// NSF_CL_59992103190701
// MI455X (gfx1250) — matrix operand order repaired and checked by reading, not compiled, not run on hardware
//
#include <hip/hip_runtime.h>

// Problem constants (match reference)
#define BATCH 4096
#define DIMF  1024
#define HALFD 512
#define HID   1024
#define KNOTS 5
#define OUTW  7168          // HALF * (3K-1) = 512*14
#define BBND  3.0f

// GEMM tiling
#define BM 128
#define BN 128
#define BK 32

#define GLOBAL_AS __attribute__((address_space(1)))
#define LDS_AS    __attribute__((address_space(3)))

#if __has_builtin(__builtin_amdgcn_global_load_async_to_lds_b128)
#define NSF_ASYNC 1
#else
#define NSF_ASYNC 0
#endif

typedef __attribute__((ext_vector_type(16))) __bf16 v16bf;
typedef __attribute__((ext_vector_type(8)))  float  v8f;
typedef __attribute__((ext_vector_type(4)))  int    v4i;

union FragBF { uint4 u[2]; v16bf v; };

__device__ __forceinline__ unsigned short f32_to_bf16(float f) {
  unsigned int u = __float_as_uint(f);
  u += 0x7FFFu + ((u >> 16) & 1u);          // round-to-nearest-even
  return (unsigned short)(u >> 16);
}

__device__ __forceinline__ void nsf_wait_async0() {
#if __has_builtin(__builtin_amdgcn_s_wait_asynccnt)
  __builtin_amdgcn_s_wait_asynccnt(0);
#else
  asm volatile("s_wait_asynccnt 0x0" ::: "memory");
#endif
}

// ---------------------------------------------------------------------------
// fp32 (possibly strided rows) -> bf16 contiguous
// ---------------------------------------------------------------------------
__global__ __launch_bounds__(256)
void nsf_cvt_bf16(const float* __restrict__ src, int ld, int rows, int cols,
                  unsigned short* __restrict__ dst) {
  int idx = blockIdx.x * blockDim.x + threadIdx.x;
  int total = rows * cols;
  if (idx >= total) return;
  int r = idx / cols;
  int c = idx - r * cols;
  dst[idx] = f32_to_bf16(src[(size_t)r * ld + c]);
}

// ---------------------------------------------------------------------------
// BF16 WMMA GEMM:  C = act(A @ W + bias)
//   A: M x K bf16 row-major, W: K x N bf16 row-major, bias: N f32
//   MODE 0: tanh -> bf16 output     MODE 1: raw f32 output
// Block: 256 threads (8 waves, 2x4), block tile 128x128, K-step 32.
// Double-buffered LDS fed by async global->LDS copies (ASYNCcnt) when the
// toolchain exposes the builtin; otherwise a register-staged pipeline.
// ---------------------------------------------------------------------------
template <int MODE>
__global__ __launch_bounds__(256)
void nsf_gemm_bf16(const unsigned short* __restrict__ A,
                   const unsigned short* __restrict__ W,
                   const float* __restrict__ bias,
                   unsigned short* __restrict__ outBf,
                   float* __restrict__ outF,
                   int M, int N, int K) {
  __shared__ unsigned short As[2][BM * BK];   // 128 x 32 (8 KB per buffer)
  __shared__ unsigned short Bs[2][BK * BN];   // 32 x 128

  const int tid  = threadIdx.x;
  const int lane = tid & 31;
  const int wave = tid >> 5;
  const int wm   = wave >> 2;     // 0..1  -> 64-row strip
  const int wn   = wave & 3;      // 0..3  -> 32-col strip
  const int bM   = blockIdx.y * BM;
  const int bN   = blockIdx.x * BN;

  // global->LDS staging: A tile 8KB, B tile 8KB; 32B per thread each
  const int arow = tid >> 1;            // 0..127
  const int acol = (tid & 1) * 16;      // 0 or 16
  const int brow = tid >> 3;            // 0..31
  const int bcol = (tid & 7) * 16;      // 0..112

  const int m16  = lane & 15;
  const int half = lane >> 4;

  const unsigned short* gA0 = &A[(size_t)(bM + arow) * K + acol];
  const unsigned short* gB0 = &W[(size_t)brow * N + (bN + bcol)];

  v8f acc[4][2] = {};
  const int ksteps = K / BK;

#if NSF_ASYNC
  auto stage = [&](int buf, int k0) {
    __builtin_amdgcn_global_load_async_to_lds_b128(
        (GLOBAL_AS v4i*)(gA0 + k0),
        (LDS_AS v4i*)&As[buf][arow * BK + acol], 0, 0);
    __builtin_amdgcn_global_load_async_to_lds_b128(
        (GLOBAL_AS v4i*)(gA0 + k0 + 8),
        (LDS_AS v4i*)&As[buf][arow * BK + acol + 8], 0, 0);
    __builtin_amdgcn_global_load_async_to_lds_b128(
        (GLOBAL_AS v4i*)(gB0 + (size_t)k0 * N),
        (LDS_AS v4i*)&Bs[buf][brow * BN + bcol], 0, 0);
    __builtin_amdgcn_global_load_async_to_lds_b128(
        (GLOBAL_AS v4i*)(gB0 + (size_t)k0 * N + 8),
        (LDS_AS v4i*)&Bs[buf][brow * BN + bcol + 8], 0, 0);
  };
#else
  uint4 ra[2], rb[2];
  auto gload = [&](int k0) {
    const uint4* ga = reinterpret_cast<const uint4*>(gA0 + k0);
    ra[0] = ga[0];
    ra[1] = ga[1];
    const uint4* gb = reinterpret_cast<const uint4*>(gB0 + (size_t)k0 * N);
    rb[0] = gb[0];
    rb[1] = gb[1];
  };
  auto sstore = [&](int buf) {
    *reinterpret_cast<uint4*>(&As[buf][arow * BK + acol])     = ra[0];
    *reinterpret_cast<uint4*>(&As[buf][arow * BK + acol + 8]) = ra[1];
    *reinterpret_cast<uint4*>(&Bs[buf][brow * BN + bcol])     = rb[0];
    *reinterpret_cast<uint4*>(&Bs[buf][brow * BN + bcol + 8]) = rb[1];
  };
#endif

  auto compute = [&](int buf) {
    // A fragments: lane holds row (lane%16); K chunks half*8 and 16+half*8
    FragBF af[4];
#pragma unroll
    for (int ms = 0; ms < 4; ++ms) {
      const int r = wm * 64 + ms * 16 + m16;
      af[ms].u[0] = *reinterpret_cast<const uint4*>(&As[buf][r * BK + half * 8]);
      af[ms].u[1] =
          *reinterpret_cast<const uint4*>(&As[buf][r * BK + 16 + half * 8]);
    }
    // B fragments: lane holds column N=lane%16; K chunks half*8 and 16+half*8
    FragBF bfm[2];
#pragma unroll
    for (int ns = 0; ns < 2; ++ns) {
      const int c = wn * 32 + ns * 16;
      unsigned short* bp = reinterpret_cast<unsigned short*>(bfm[ns].u);
#pragma unroll
      for (int s = 0; s < 16; ++s) {
        const int kk = (s < 8) ? (half * 8 + s) : (16 + half * 8 + (s - 8));
        bp[s] = Bs[buf][kk * BN + c + m16];
      }
    }
#pragma unroll
    for (int ms = 0; ms < 4; ++ms)
#pragma unroll
      for (int ns = 0; ns < 2; ++ns)
        acc[ms][ns] = __builtin_amdgcn_wmma_f32_16x16x32_bf16(
            false, af[ms].v, false, bfm[ns].v, (short)0, acc[ms][ns],
            false, false);
  };

#if NSF_ASYNC
  stage(0, 0);
  nsf_wait_async0();
  __syncthreads();
  for (int kt = 0; kt < ksteps; ++kt) {
    const int buf = kt & 1;
    if (kt + 1 < ksteps) stage(buf ^ 1, (kt + 1) * BK);  // prefetch next tile
    compute(buf);
    if (kt + 1 < ksteps) nsf_wait_async0();  // next tile landed (this wave)
    __syncthreads();                         // ...and is visible block-wide
  }
#else
  gload(0);
  sstore(0);
  if (ksteps > 1) gload(BK);
  __syncthreads();
  for (int kt = 0; kt < ksteps; ++kt) {
    const int buf = kt & 1;
    compute(buf);
    if (kt + 1 < ksteps) {
      __syncthreads();                       // all waves done reading buf^1
      sstore(buf ^ 1);
      if (kt + 2 < ksteps) gload((kt + 2) * BK);
      __syncthreads();
    }
  }
#endif

  // epilogue: C layout -> N = lane%16, M = v + 8*(lane/16)
#pragma unroll
  for (int ms = 0; ms < 4; ++ms) {
    const int row0 = bM + wm * 64 + ms * 16 + half * 8;
#pragma unroll
    for (int ns = 0; ns < 2; ++ns) {
      const int col = bN + wn * 32 + ns * 16 + m16;
      const float bv = bias[col];
#pragma unroll
      for (int v = 0; v < 8; ++v) {
        float r = acc[ms][ns][v] + bv;
        if (MODE == 0) {
          outBf[(size_t)(row0 + v) * N + col] = f32_to_bf16(tanhf(r));
        } else {
          outF[(size_t)(row0 + v) * N + col] = r;
        }
      }
    }
  }
}

// ---------------------------------------------------------------------------
// Rational-quadratic spline transform + per-row log-det reduction.
// One block per batch row; 256 threads cover 512 columns.
// ---------------------------------------------------------------------------
__global__ __launch_bounds__(256)
void nsf_spline(const float* __restrict__ xin, const float* __restrict__ net,
                float* __restrict__ zout, unsigned short* __restrict__ ybf,
                float* __restrict__ logdet, int accumulate) {
  const int b = blockIdx.x;
  __shared__ float red[256];
  float lacc = 0.f;

  for (int j = threadIdx.x; j < HALFD; j += 256) {
    const float x = xin[(size_t)b * DIMF + j];
    const float* o = net + (size_t)b * OUTW + (size_t)j * (3 * KNOTS - 1);

    float Wv[KNOTS], Hv[KNOTS], dkn[KNOTS + 1];
    float mw = o[0], mh = o[KNOTS];
#pragma unroll
    for (int i = 1; i < KNOTS; ++i) {
      mw = fmaxf(mw, o[i]);
      mh = fmaxf(mh, o[KNOTS + i]);
    }
    float sw = 0.f, sh = 0.f;
#pragma unroll
    for (int i = 0; i < KNOTS; ++i) {
      Wv[i] = expf(o[i] - mw);         sw += Wv[i];
      Hv[i] = expf(o[KNOTS + i] - mh); sh += Hv[i];
    }
    const float fw = 2.f * BBND / sw, fh = 2.f * BBND / sh;
#pragma unroll
    for (int i = 0; i < KNOTS; ++i) { Wv[i] *= fw; Hv[i] *= fh; }

    float cw[KNOTS + 1], ch[KNOTS + 1];
    cw[0] = -BBND; ch[0] = -BBND;
#pragma unroll
    for (int i = 0; i < KNOTS; ++i) {
      cw[i + 1] = cw[i] + Wv[i];
      ch[i + 1] = ch[i] + Hv[i];
    }
    cw[KNOTS] = BBND; ch[KNOTS] = BBND;

    dkn[0] = 1.f; dkn[KNOTS] = 1.f;
#pragma unroll
    for (int i = 0; i < KNOTS - 1; ++i) {
      const float dp = o[2 * KNOTS + i];
      dkn[i + 1] = (dp > 20.f) ? dp : log1pf(expf(dp));   // softplus
    }

    const bool inside = (x >= -BBND) && (x <= BBND);
    const float xc = fminf(fmaxf(x, -BBND), BBND);
    int idx = -1;
#pragma unroll
    for (int i = 0; i < KNOTS; ++i) idx += (xc >= cw[i]) ? 1 : 0;
    idx = idx < 0 ? 0 : (idx > KNOTS - 1 ? KNOTS - 1 : idx);

    const float wk = Wv[idx], hk = Hv[idx];
    const float cwk = cw[idx], chk = ch[idx];
    const float dk = dkn[idx], dk1 = dkn[idx + 1];
    const float s    = hk / wk;
    const float th   = (xc - cwk) / wk;
    const float th1m = th * (1.f - th);
    const float den  = s + (dk1 + dk - 2.f * s) * th1m;
    const float y    = chk + hk * (s * th * th + dk * th1m) / den;
    const float omt  = 1.f - th;
    const float ld   = logf(s * s * (dk1 * th * th + 2.f * s * th1m +
                                     dk * omt * omt)) - 2.f * logf(den);

    const float yo = inside ? y : x;
    zout[(size_t)b * DIMF + j] = yo;
    if (ybf) ybf[(size_t)b * HALFD + j] = f32_to_bf16(yo);
    lacc += inside ? ld : 0.f;
  }

  red[threadIdx.x] = lacc;
  __syncthreads();
#pragma unroll
  for (int s = 128; s > 0; s >>= 1) {
    if ((int)threadIdx.x < s) red[threadIdx.x] += red[threadIdx.x + s];
    __syncthreads();
  }
  if (threadIdx.x == 0) {
    if (accumulate) logdet[b] += red[0];
    else            logdet[b]  = red[0];
  }
}

// ---------------------------------------------------------------------------
extern "C" void kernel_launch(void* const* d_in, const int* in_sizes, int n_in,
                              void* d_out, int out_size, void* d_ws, size_t ws_size,
                              hipStream_t stream) {
  (void)in_sizes; (void)n_in; (void)out_size; (void)ws_size;

  const float* x = (const float*)d_in[0];
  const float* W1[2] = {(const float*)d_in[1], (const float*)d_in[7]};
  const float* b1[2] = {(const float*)d_in[2], (const float*)d_in[8]};
  const float* W2[2] = {(const float*)d_in[3], (const float*)d_in[9]};
  const float* b2[2] = {(const float*)d_in[4], (const float*)d_in[10]};
  const float* W3[2] = {(const float*)d_in[5], (const float*)d_in[11]};
  const float* b3[2] = {(const float*)d_in[6], (const float*)d_in[12]};

  char* ws = (char*)d_ws;
  size_t off = 0;
  auto take = [&](size_t bytes) -> char* {
    char* p = ws + off;
    off = (off + bytes + 255) & ~(size_t)255;
    return p;
  };

  unsigned short *w1bf[2], *w2bf[2], *w3bf[2];
  for (int c = 0; c < 2; ++c) {
    w1bf[c] = (unsigned short*)take((size_t)HALFD * HID * 2);
    w2bf[c] = (unsigned short*)take((size_t)HID * HID * 2);
    w3bf[c] = (unsigned short*)take((size_t)HID * OUTW * 2);
  }
  unsigned short* xin_bf = (unsigned short*)take((size_t)BATCH * HALFD * 2);
  unsigned short* h1bf   = (unsigned short*)take((size_t)BATCH * HID * 2);
  unsigned short* h2bf   = (unsigned short*)take((size_t)BATCH * HID * 2);
  float*          netf   = (float*)take((size_t)BATCH * OUTW * 4);

  float* z = (float*)d_out;
  float* logdet = z + (size_t)BATCH * DIMF;

  const dim3 blk(256);
  auto cvt = [&](const float* src, int ld, int rows, int cols,
                 unsigned short* dst) {
    const int total = rows * cols;
    nsf_cvt_bf16<<<dim3((total + 255) / 256), blk, 0, stream>>>(src, ld, rows,
                                                                cols, dst);
  };

  // weights -> bf16
  for (int c = 0; c < 2; ++c) {
    cvt(W1[c], HID, HALFD, HID, w1bf[c]);
    cvt(W2[c], HID, HID, HID, w2bf[c]);
    cvt(W3[c], OUTW, HID, OUTW, w3bf[c]);
  }
  // x0 (cols 0..511 of x) -> bf16, first coupling's FCNN input
  cvt(x, DIMF, BATCH, HALFD, xin_bf);

  for (int c = 0; c < 2; ++c) {
    nsf_gemm_bf16<0><<<dim3(HID / BN, BATCH / BM), blk, 0, stream>>>(
        xin_bf, w1bf[c], b1[c], h1bf, nullptr, BATCH, HID, HALFD);
    nsf_gemm_bf16<0><<<dim3(HID / BN, BATCH / BM), blk, 0, stream>>>(
        h1bf, w2bf[c], b2[c], h2bf, nullptr, BATCH, HID, HID);
    nsf_gemm_bf16<1><<<dim3(OUTW / BN, BATCH / BM), blk, 0, stream>>>(
        h2bf, w3bf[c], b3[c], nullptr, netf, BATCH, OUTW, HID);

    // coupling 0 transforms x1 (cols 512..), coupling 1 transforms x0
    const float* xs = x + (c == 0 ? HALFD : 0);
    float* zs = z + (c == 0 ? HALFD : 0);
    // transformed x1 (bf16) becomes the second coupling's FCNN input
    nsf_spline<<<dim3(BATCH), blk, 0, stream>>>(
        xs, netf, zs, (c == 0) ? xin_bf : nullptr, logdet, c);
  }
}
